// CrossAttentionForQA_31808527794538
// MI455X (gfx1250) — compile-verified, matrix operand order checked by reading, not run on hardware
//
#include <hip/hip_runtime.h>
#include <hip/hip_bf16.h>

typedef __bf16 bf16;
typedef __attribute__((ext_vector_type(8)))  float v8f;
typedef __attribute__((ext_vector_type(16))) bf16  v16bf;
typedef __attribute__((ext_vector_type(8)))  bf16  v8bf;

constexpr int kB = 8, kS = 1024, kE = 1024, kH = 16, kHD = 64, kF = 4096;
constexpr int kM = kB * kS;  // 8192 token rows
constexpr float kNEG = -1.0e30f;

// ---------------------------------------------------------------------------
// WMMA + async-copy helpers (CDNA5 wave32)
// ---------------------------------------------------------------------------
__device__ __forceinline__ v8f wmma_bf16(v16bf a, v16bf b, v8f c) {
  return __builtin_amdgcn_wmma_f32_16x16x32_bf16(
      /*neg_a=*/false, a, /*neg_b=*/false, b,
      /*c_mod=*/(short)0, c, /*reuse_a=*/false, /*reuse_b=*/false);
}

// A/B fragment for one lane: two contiguous 8-half runs at p and p+16.
// (16-bit operand layout: lanes<16 hold K = 0..7 & 16..23 of the 32-K tile,
//  lanes>=16 hold K = 8..15 & 24..31 — caller passes p already offset by hi*8.)
__device__ __forceinline__ v16bf ld_frag(const bf16* p) {
  union { v16bf v; v8bf h[2]; } u;
  u.h[0] = *(const v8bf*)(p);
  u.h[1] = *(const v8bf*)(p + 16);
  return u.v;
}

// 16-byte global -> LDS async copy (ASYNCcnt-tracked, no VGPR staging).
// LDS destination = low 32 bits of the generic pointer (LDS aperture offset).
__device__ __forceinline__ void async_copy16(const bf16* gsrc, bf16* ldst) {
  unsigned lds = (unsigned)(size_t)ldst;
  asm volatile("global_load_async_to_lds_b128 %0, %1, off"
               :: "v"(lds), "v"(gsrc) : "memory");
}
__device__ __forceinline__ void wait_async0() {
  asm volatile("s_wait_asynccnt 0x0" ::: "memory");
}

// ---------------------------------------------------------------------------
// Small conversion kernels
// ---------------------------------------------------------------------------
__global__ __launch_bounds__(256) void cvt_bf16_kernel(const float* __restrict__ in,
                                                       bf16* __restrict__ out, int n) {
  int idx = (blockIdx.x * 256 + threadIdx.x) * 4;
  if (idx < n) {
    float4 v = *(const float4*)(in + idx);
    out[idx + 0] = (bf16)v.x;
    out[idx + 1] = (bf16)v.y;
    out[idx + 2] = (bf16)v.z;
    out[idx + 3] = (bf16)v.w;
  }
}

// in: [K,N] fp32 row-major  ->  out: [N,K] bf16 row-major
__global__ __launch_bounds__(256) void transpose_cvt_kernel(const float* __restrict__ in,
                                                            bf16* __restrict__ out,
                                                            int K, int N) {
  size_t idx = (size_t)blockIdx.x * 256 + threadIdx.x;
  if (idx < (size_t)K * N) {
    int n = (int)(idx / K), k = (int)(idx % K);
    out[idx] = (bf16)in[(size_t)k * N + n];
  }
}

// ---------------------------------------------------------------------------
// Generic bf16 GEMM: C[M,N] = A[M,K] * Bt[N,K]^T + bias, fused epilogues.
// Block = 256 threads (8 wave32), tile 128(M) x 64(N), waves 4x2, wave 32x32.
// Double-buffered LDS filled by GLOBAL_LOAD_ASYNC_TO_LDS_B128; one barrier
// per K-step; async fill of tile t+1 overlaps the 4 WMMAs of tile t.
// ---------------------------------------------------------------------------
enum { EPI_PLAIN = 0, EPI_GELU = 1, EPI_QKV = 2 };

template <int EPI>
__global__ __launch_bounds__(256) void gemm_bf16_kernel(
    const bf16* __restrict__ A, const bf16* __restrict__ Bt,
    const float* __restrict__ bias,
    float* __restrict__ Cf, bf16* __restrict__ Cb,
    bf16* __restrict__ qO, bf16* __restrict__ kO, bf16* __restrict__ vtO,
    int M, int N, int K) {
  __shared__ __attribute__((aligned(16))) bf16 sA[2][128 * 32];
  __shared__ __attribute__((aligned(16))) bf16 sB[2][64 * 32];
  const int tid = threadIdx.x;
  const int lane = tid & 31, wave = tid >> 5;
  const int hi = lane >> 4, lm = lane & 15;
  const int waveM = wave & 3, waveN = wave >> 2;
  const int m0 = blockIdx.y * 128;
  const int n0 = blockIdx.x * 64;

  // Per-thread fixed staging slots (chunks of 8 halves = 16 B).
  const int aRow0 = (tid * 2) >> 2,     aCh0 = (tid * 2) & 3;
  const int aRow1 = (tid * 2 + 1) >> 2, aCh1 = (tid * 2 + 1) & 3;
  const int bRow  = tid >> 2,           bCh  = tid & 3;

  auto stage = [&](int kt, int buf) {
    const int kk = kt * 32;
    async_copy16(&A[(size_t)(m0 + aRow0) * K + kk + aCh0 * 8],
                 &sA[buf][aRow0 * 32 + aCh0 * 8]);
    async_copy16(&A[(size_t)(m0 + aRow1) * K + kk + aCh1 * 8],
                 &sA[buf][aRow1 * 32 + aCh1 * 8]);
    async_copy16(&Bt[(size_t)(n0 + bRow) * K + kk + bCh * 8],
                 &sB[buf][bRow * 32 + bCh * 8]);
  };

  v8f acc[2][2] = {};
  const int kTiles = K >> 5;

  stage(0, 0);
  wait_async0();
  __syncthreads();

  for (int kt = 0; kt < kTiles; ++kt) {
    const int cur = kt & 1;
    if (kt + 1 < kTiles) stage(kt + 1, cur ^ 1);

    v16bf af[2], bfr[2];
#pragma unroll
    for (int mt = 0; mt < 2; ++mt)
      af[mt] = ld_frag(&sA[cur][(waveM * 32 + mt * 16 + lm) * 32 + hi * 8]);
#pragma unroll
    for (int nt = 0; nt < 2; ++nt)
      bfr[nt] = ld_frag(&sB[cur][(waveN * 32 + nt * 16 + lm) * 32 + hi * 8]);
#pragma unroll
    for (int mt = 0; mt < 2; ++mt)
#pragma unroll
      for (int nt = 0; nt < 2; ++nt)
        acc[mt][nt] = wmma_bf16(af[mt], bfr[nt], acc[mt][nt]);

    wait_async0();    // next-buffer fill complete (overlapped with WMMAs)
    __syncthreads();  // all waves done reading `cur` before it is refilled
  }

#pragma unroll
  for (int mt = 0; mt < 2; ++mt) {
#pragma unroll
    for (int nt = 0; nt < 2; ++nt) {
#pragma unroll
      for (int i = 0; i < 8; ++i) {
        int row = m0 + waveM * 32 + mt * 16 + hi * 8 + i;
        int col = n0 + waveN * 32 + nt * 16 + lm;
        float v = acc[mt][nt][i] + bias[col];
        if constexpr (EPI == EPI_PLAIN) {
          Cf[(size_t)row * N + col] = v;
        } else if constexpr (EPI == EPI_GELU) {
          float g = 0.5f * v * (1.0f + erff(v * 0.70710678118654752f));
          Cb[(size_t)row * N + col] = (bf16)g;
        } else {  // EPI_QKV: scatter to Q/K [b,h,s,d] and V^T [b,h,d,s]
          int bb = row >> 10, s = row & 1023;
          int sec = col >> 10, cc = col & 1023;
          int h = cc >> 6, d = cc & 63;
          size_t qkIdx = ((size_t)(bb * kH + h) * kS + s) * kHD + d;
          if (sec == 0)      qO[qkIdx] = (bf16)v;
          else if (sec == 1) kO[qkIdx] = (bf16)v;
          else vtO[((size_t)(bb * kH + h) * kHD + d) * kS + s] = (bf16)v;
        }
      }
    }
  }
}

// ---------------------------------------------------------------------------
// Flash attention: block = (b, h, 128 q rows), 8 waves x 16 q rows each.
// Key tiles of 32, double-buffered + async-filled; online softmax; WMMA bf16.
// ---------------------------------------------------------------------------
__global__ __launch_bounds__(256) void attention_kernel(
    const bf16* __restrict__ Q, const bf16* __restrict__ Kk,
    const bf16* __restrict__ VT,
    const int* __restrict__ am, const int* __restrict__ tt,
    bf16* __restrict__ ctx) {
  __shared__ __attribute__((aligned(16))) bf16 sQ[128 * 64];
  __shared__ __attribute__((aligned(16))) bf16 sK[2][32 * 64];
  __shared__ __attribute__((aligned(16))) bf16 sVT[2][64 * 32];
  __shared__ __attribute__((aligned(16))) bf16 sP[8 * 16 * 32];  // per-wave 16x32
  __shared__ float sAmK[2][32];
  __shared__ float sMaskK[2][32];

  const int tid = threadIdx.x;
  const int lane = tid & 31, w = tid >> 5;
  const int hi = lane >> 4, lm = lane & 15;
  const int qt = blockIdx.x, h = blockIdx.y, b = blockIdx.z;
  const size_t headBase = (size_t)(b * kH + h) * kS * kHD;
  const size_t vtBase   = (size_t)(b * kH + h) * kHD * kS;

  // Stage Q tile (contiguous 128*64 halves) via async copies.
#pragma unroll
  for (int j = 0; j < 4; ++j) {
    int idx = tid * 4 + j;  // 0..1023 chunks of 8
    async_copy16(&Q[headBase + (size_t)qt * 128 * kHD + idx * 8], &sQ[idx * 8]);
  }

  auto stageKV = [&](int kt, int buf) {
    async_copy16(&Kk[headBase + (size_t)kt * 32 * kHD + tid * 8],
                 &sK[buf][tid * 8]);
    int d = tid >> 2, c = tid & 3;
    async_copy16(&VT[vtBase + (size_t)d * kS + kt * 32 + c * 8],
                 &sVT[buf][d * 32 + c * 8]);
    if (tid < 32) {
      int col = kt * 32 + tid;
      int a2 = am[b * kS + col];
      int t2 = tt[b * kS + col];
      sAmK[buf][tid]   = (a2 != 0) ? 1.0f : 0.0f;
      sMaskK[buf][tid] = ((t2 == 1) || (a2 == 0) || (col == 0)) ? 1.0f : 0.0f;
    }
  };

  float aq[8];      // query-row activity (for additive 0/1 mask)
  float mrow[8], lrow[8];
#pragma unroll
  for (int i = 0; i < 8; ++i) {
    int r = qt * 128 + w * 16 + hi * 8 + i;
    aq[i] = (am[b * kS + r] != 0) ? 1.0f : 0.0f;
    mrow[i] = -3.0e38f;
    lrow[i] = 0.0f;
  }
  v8f o[4] = {};

  constexpr int kTiles = kS / 32;
  stageKV(0, 0);
  wait_async0();
  __syncthreads();

  for (int kt = 0; kt < kTiles; ++kt) {
    const int cur = kt & 1;
    if (kt + 1 < kTiles) stageKV(kt + 1, cur ^ 1);

    // Scores: 16 q rows x 32 keys (2 n-tiles, 2 chained WMMAs over HD=64)
    v16bf a0 = ld_frag(&sQ[(w * 16 + lm) * 64 + hi * 8]);
    v16bf a1 = ld_frag(&sQ[(w * 16 + lm) * 64 + 32 + hi * 8]);
    v8f s[2] = {};
#pragma unroll
    for (int nt = 0; nt < 2; ++nt) {
      v16bf b0 = ld_frag(&sK[cur][(nt * 16 + lm) * 64 + hi * 8]);
      v16bf b1 = ld_frag(&sK[cur][(nt * 16 + lm) * 64 + 32 + hi * 8]);
      s[nt] = wmma_bf16(a0, b0, s[nt]);
      s[nt] = wmma_bf16(a1, b1, s[nt]);
    }

    float amk0 = sAmK[cur][lm],   amk1 = sAmK[cur][16 + lm];
    float mk0  = sMaskK[cur][lm], mk1  = sMaskK[cur][16 + lm];
#pragma unroll
    for (int i = 0; i < 8; ++i) {
      float v0 = s[0][i] * 0.125f + aq[i] * amk0;   // 1/sqrt(64) + additive 0/1
      float v1 = s[1][i] * 0.125f + aq[i] * amk1;
      if (mk0 > 0.5f) v0 = kNEG;
      if (mk1 > 0.5f) v1 = kNEG;
      // row max over 32 keys (16-lane xor reduction; rows live in lane halves)
      float rm = fmaxf(v0, v1);
      rm = fmaxf(rm, __shfl_xor(rm, 1, 32));
      rm = fmaxf(rm, __shfl_xor(rm, 2, 32));
      rm = fmaxf(rm, __shfl_xor(rm, 4, 32));
      rm = fmaxf(rm, __shfl_xor(rm, 8, 32));
      float nm = fmaxf(mrow[i], rm);
      float sc = __expf(mrow[i] - nm);
      float p0 = __expf(v0 - nm);
      float p1 = __expf(v1 - nm);
      float rs = p0 + p1;
      rs += __shfl_xor(rs, 1, 32);
      rs += __shfl_xor(rs, 2, 32);
      rs += __shfl_xor(rs, 4, 32);
      rs += __shfl_xor(rs, 8, 32);
      lrow[i] = lrow[i] * sc + rs;
      mrow[i] = nm;
#pragma unroll
      for (int ot = 0; ot < 4; ++ot) o[ot][i] = o[ot][i] * sc;
      // stage P into A-fragment layout (per-wave region, wave-internal RAW)
      sP[(w * 16 + hi * 8 + i) * 32 + lm]      = (bf16)p0;
      sP[(w * 16 + hi * 8 + i) * 32 + 16 + lm] = (bf16)p1;
    }

    // O += P(16x32) @ V(32x64): 4 n-tiles of 16 dims
    v16bf pa = ld_frag(&sP[(w * 16 + lm) * 32 + hi * 8]);
#pragma unroll
    for (int ot = 0; ot < 4; ++ot) {
      v16bf bv = ld_frag(&sVT[cur][(ot * 16 + lm) * 32 + hi * 8]);
      o[ot] = wmma_bf16(pa, bv, o[ot]);
    }

    wait_async0();    // next K/V buffer filled (overlapped with compute)
    __syncthreads();  // all waves done with `cur` before it is refilled
  }

  // ctx[b, s, h*64+d] in bf16 for the out-projection GEMM
#pragma unroll
  for (int ot = 0; ot < 4; ++ot) {
#pragma unroll
    for (int i = 0; i < 8; ++i) {
      int qrow = qt * 128 + w * 16 + hi * 8 + i;
      int d = ot * 16 + lm;
      float v = o[ot][i] / lrow[i];
      ctx[((size_t)b * kS + qrow) * kE + h * kHD + d] = (bf16)v;
    }
  }
}

// ---------------------------------------------------------------------------
// out = LayerNorm(a + r) * g + be ; optionally also emit bf16 copy
// One block (256 threads) per token row of E=1024.
// ---------------------------------------------------------------------------
__global__ __launch_bounds__(256) void add_ln_kernel(
    const float* __restrict__ a, const float* __restrict__ r,
    const float* __restrict__ g, const float* __restrict__ be,
    float* __restrict__ outF, bf16* __restrict__ outB) {
  __shared__ float red[256];
  const int row = blockIdx.x, t = threadIdx.x;
  const size_t base = (size_t)row * kE;
  float4 va = ((const float4*)(a + base))[t];
  float4 vr = ((const float4*)(r + base))[t];
  float x0 = va.x + vr.x, x1 = va.y + vr.y, x2 = va.z + vr.z, x3 = va.w + vr.w;

  red[t] = x0 + x1 + x2 + x3;
  __syncthreads();
  for (int o = 128; o > 0; o >>= 1) { if (t < o) red[t] += red[t + o]; __syncthreads(); }
  float mu = red[0] * (1.0f / kE);
  __syncthreads();

  float d0 = x0 - mu, d1 = x1 - mu, d2 = x2 - mu, d3 = x3 - mu;
  red[t] = d0 * d0 + d1 * d1 + d2 * d2 + d3 * d3;
  __syncthreads();
  for (int o = 128; o > 0; o >>= 1) { if (t < o) red[t] += red[t + o]; __syncthreads(); }
  float inv = rsqrtf(red[0] * (1.0f / kE) + 1e-12f);

  float4 gw = ((const float4*)g)[t];
  float4 bb = ((const float4*)be)[t];
  float o0 = d0 * inv * gw.x + bb.x;
  float o1 = d1 * inv * gw.y + bb.y;
  float o2 = d2 * inv * gw.z + bb.z;
  float o3 = d3 * inv * gw.w + bb.w;
  ((float4*)(outF + base))[t] = make_float4(o0, o1, o2, o3);
  if (outB) {
    outB[base + t * 4 + 0] = (bf16)o0;
    outB[base + t * 4 + 1] = (bf16)o1;
    outB[base + t * 4 + 2] = (bf16)o2;
    outB[base + t * 4 + 3] = (bf16)o3;
  }
}

// ---------------------------------------------------------------------------
extern "C" void kernel_launch(void* const* d_in, const int* in_sizes, int n_in,
                              void* d_out, int out_size, void* d_ws, size_t ws_size,
                              hipStream_t stream) {
  const float* x    = (const float*)d_in[0];
  const int*   am   = (const int*)d_in[1];
  const int*   tt   = (const int*)d_in[2];
  const float* wqkv = (const float*)d_in[3];
  const float* bqkv = (const float*)d_in[4];
  const float* wout = (const float*)d_in[5];
  const float* bout = (const float*)d_in[6];
  const float* lnw  = (const float*)d_in[7];
  const float* lnb  = (const float*)d_in[8];
  const float* win  = (const float*)d_in[9];
  const float* bin  = (const float*)d_in[10];
  const float* wo2  = (const float*)d_in[11];
  const float* bo2  = (const float*)d_in[12];
  float* out = (float*)d_out;

  char* ws = (char*)d_ws;
  size_t off = 0;
  auto take = [&](size_t bytes) -> char* {
    char* p = ws + off;
    off += (bytes + 255) & ~(size_t)255;
    return p;
  };
  bf16*  xb     = (bf16*)take((size_t)kM * kE * 2);          // 16 MB
  bf16*  wqkvT  = (bf16*)take((size_t)3 * kE * kE * 2);      //  6 MB
  bf16*  woutT  = (bf16*)take((size_t)kE * kE * 2);          //  2 MB
  bf16*  winT   = (bf16*)take((size_t)kF * kE * 2);          //  8 MB
  bf16*  wo2T   = (bf16*)take((size_t)kE * kF * 2);          //  8 MB
  bf16*  qBuf   = (bf16*)take((size_t)kM * kE * 2);          // 16 MB  [b,h,s,d]
  bf16*  kBuf   = (bf16*)take((size_t)kM * kE * 2);          // 16 MB  [b,h,s,d]
  bf16*  vtBuf  = (bf16*)take((size_t)kM * kE * 2);          // 16 MB  [b,h,d,s]
  bf16*  ctx    = (bf16*)take((size_t)kM * kE * 2);          // 16 MB  [b,s,e]
  float* attnF  = (float*)take((size_t)kM * kE * 4);         // 32 MB (reused as out2)
  float* hF     = (float*)take((size_t)kM * kE * 4);         // 32 MB
  bf16*  hB     = (bf16*)take((size_t)kM * kE * 2);          // 16 MB
  bf16*  ff     = qBuf;  // 64 MB alias over q/k/vT/ctx (dead after out-proj)

  // 1) bf16 conversions / weight transposes
  cvt_bf16_kernel<<<(kM * kE) / 1024, 256, 0, stream>>>(x, xb, kM * kE);
  transpose_cvt_kernel<<<(3 * kE * kE) / 256, 256, 0, stream>>>(wqkv, wqkvT, kE, 3 * kE);
  transpose_cvt_kernel<<<(kE * kE) / 256, 256, 0, stream>>>(wout, woutT, kE, kE);
  transpose_cvt_kernel<<<(kE * kF) / 256, 256, 0, stream>>>(win, winT, kE, kF);
  transpose_cvt_kernel<<<(kF * kE) / 256, 256, 0, stream>>>(wo2, wo2T, kF, kE);

  // 2) QKV projection with head-split scatter epilogue
  gemm_bf16_kernel<EPI_QKV><<<dim3(3 * kE / 64, kM / 128), 256, 0, stream>>>(
      xb, wqkvT, bqkv, nullptr, nullptr, qBuf, kBuf, vtBuf, kM, 3 * kE, kE);

  // 3) Flash attention
  attention_kernel<<<dim3(kS / 128, kH, kB), 256, 0, stream>>>(
      qBuf, kBuf, vtBuf, am, tt, ctx);

  // 4) Output projection
  gemm_bf16_kernel<EPI_PLAIN><<<dim3(kE / 64, kM / 128), 256, 0, stream>>>(
      ctx, woutT, bout, attnF, nullptr, nullptr, nullptr, nullptr, kM, kE, kE);

  // 5) h = LN(attn_out + x)
  add_ln_kernel<<<kM, 256, 0, stream>>>(attnF, x, lnw, lnb, hF, hB);

  // 6) FFN up + exact GELU (bf16 out)
  gemm_bf16_kernel<EPI_GELU><<<dim3(kF / 64, kM / 128), 256, 0, stream>>>(
      hB, winT, bin, nullptr, ff, nullptr, nullptr, nullptr, kM, kF, kE);

  // 7) FFN down (f32 out, reusing attnF)
  gemm_bf16_kernel<EPI_PLAIN><<<dim3(kE / 64, kM / 128), 256, 0, stream>>>(
      ff, wo2T, bo2, attnF, nullptr, nullptr, nullptr, nullptr, kM, kE, kF);

  // 8) out = LN(ffn_out + h)
  add_ln_kernel<<<kM, 256, 0, stream>>>(attnF, hF, lnw, lnb, out, nullptr);

  (void)in_sizes; (void)n_in; (void)out_size; (void)ws_size;
}
